// RoIHeads_85203561218861
// MI455X (gfx1250) — matrix operand order repaired and checked by reading, not compiled, not run on hardware
//
#include <hip/hip_runtime.h>
#include <hip/hip_bf16.h>
#include <float.h>

// ---------------------------------------------------------------- types
typedef __attribute__((ext_vector_type(16))) __bf16 v16bf;
typedef __attribute__((ext_vector_type(8)))  float  v8f;
typedef __attribute__((ext_vector_type(4)))  int    i4_t;

// ---------------------------------------------------------------- CDNA5 async copy probe
#if __has_builtin(__builtin_amdgcn_global_load_async_to_lds_b128) && \
    __has_builtin(__builtin_amdgcn_s_wait_asynccnt)
#define HAVE_ASYNC_LDS 1
#define ASYNC_GLB(p) ((__attribute__((address_space(1))) i4_t*)(p))
#define ASYNC_LDS(p) ((__attribute__((address_space(3))) i4_t*)(p))
#else
#define HAVE_ASYNC_LDS 0
#endif

// ---------------------------------------------------------------- shapes
constexpr int Bn   = 2;
constexpr int Cc   = 256;
constexpr int Hh   = 50;
constexpr int Ww   = 76;
constexpr int Pp   = 512;
constexpr int NROI = Bn * Pp;          // 1024
constexpr int D_IN = Cc * 7 * 7;       // 12544
constexpr int REP  = 4096;
constexpr int NCLS = 81;
constexpr int FLATN = Pp * (NCLS - 1); // 40960 per image
constexpr int PRE_K = 1000;
constexpr int DETN  = 100;
constexpr float SCALE_  = 1.0f / 16.0f;
constexpr float SCORE_T = 0.05f;
constexpr float NMS_T   = 0.3f;
constexpr float BBOX_CLIP = 4.135166556742356f;

// d_out float offsets
constexpr size_t OFF_LOGITS = 0;                                   // 1024*81
constexpr size_t OFF_BBOX   = (size_t)NROI * NCLS;                 // 82944
constexpr size_t OFF_DETB   = OFF_BBOX + (size_t)NROI * NCLS * 4;  // 414720
constexpr size_t OFF_DETS   = OFF_DETB + (size_t)Bn * DETN * 4;    // 415520
constexpr size_t OFF_DETL   = OFF_DETS + (size_t)Bn * DETN;        // 415720

// workspace byte offsets (all 256B aligned)
constexpr size_t WS_X    = 0;            // 1024*12544 bf16  = 25,690,112
constexpr size_t WS_H6   = 25690112;     // 1024*4096  bf16  =  8,388,608
constexpr size_t WS_H7   = 34078720;     // 1024*4096  bf16  =  8,388,608
constexpr size_t WS_FSC  = 42467328;     // 2*40960 f32      =    327,680
constexpr size_t WS_FBX  = 42795008;     // 2*40960*4 f32    =  1,310,720
constexpr size_t WS_SELS = 44105728;     // 2*1024 f32
constexpr size_t WS_SELL = 44113920;     // 2*1024 i32
constexpr size_t WS_SELB = 44122112;     // 2*1024*4 f32
constexpr size_t WS_IOU  = 44154880;     // 2*1024*1024 f32  =  8,388,608

// ---------------------------------------------------------------- bf16 helpers (bit-exact RNE pack)
__device__ __forceinline__ unsigned short f2bfu(float f) {
    unsigned u = __builtin_bit_cast(unsigned, f);
    unsigned r = (u + 0x7FFFu + ((u >> 16) & 1u)) >> 16;
    return (unsigned short)r;
}
__device__ __forceinline__ unsigned f2bf_pk(float lo, float hi) {
    return (unsigned)f2bfu(lo) | ((unsigned)f2bfu(hi) << 16);
}

// ---------------------------------------------------------------- 1) ROI-Align -> bf16 A matrix
__global__ __launch_bounds__(256) void roi_align_kernel(
    const float* __restrict__ feats,   // (B,C,H,W)
    const float* __restrict__ props,   // (B,P,4)
    unsigned short* __restrict__ X)    // (NROI, D_IN) bf16, col = c*49 + oy*7 + ox
{
    const int n = blockIdx.x;          // roi id
    const int c = threadIdx.x;         // channel
    const int b = n / Pp;
    const float* pr = props + (size_t)n * 4;
    const float x1 = pr[0] * SCALE_, y1 = pr[1] * SCALE_;
    const float x2 = pr[2] * SCALE_, y2 = pr[3] * SCALE_;
    const float rw = fmaxf(x2 - x1, 1.0f);
    const float rh = fmaxf(y2 - y1, 1.0f);
    const float* f = feats + ((size_t)(b * Cc + c)) * Hh * Ww;

    for (int oy = 0; oy < 7; ++oy) {
        for (int ox = 0; ox < 7; ++ox) {
            float acc = 0.0f;
            #pragma unroll
            for (int sy = 0; sy < 2; ++sy) {
                #pragma unroll
                for (int sx = 0; sx < 2; ++sx) {
                    const float ty = ((float)(oy * 2 + sy) + 0.5f) * (1.0f / 14.0f);
                    const float tx = ((float)(ox * 2 + sx) + 0.5f) * (1.0f / 14.0f);
                    const float ys = y1 + ty * rh;
                    const float xs = x1 + tx * rw;
                    if (ys < -1.0f || ys > (float)Hh || xs < -1.0f || xs > (float)Ww) continue;
                    const float yc = fminf(fmaxf(ys, 0.0f), (float)(Hh - 1));
                    const float xc = fminf(fmaxf(xs, 0.0f), (float)(Ww - 1));
                    const int y0 = (int)floorf(yc);
                    const int x0 = (int)floorf(xc);
                    const int y1i = min(y0 + 1, Hh - 1);
                    const int x1i = min(x0 + 1, Ww - 1);
                    const float ly = yc - (float)y0, lx = xc - (float)x0;
                    const float hy = 1.0f - ly,      hx = 1.0f - lx;
                    acc += hy * hx * f[y0  * Ww + x0 ]
                         + hy * lx * f[y0  * Ww + x1i]
                         + ly * hx * f[y1i * Ww + x0 ]
                         + ly * lx * f[y1i * Ww + x1i];
                }
            }
            X[(size_t)n * D_IN + c * 49 + oy * 7 + ox] = f2bfu(acc * 0.25f);
        }
    }
}

// ---------------------------------------------------------------- 2) bf16 WMMA GEMM, double-buffered
//    C[M,N] = relu?(A[M,K] @ B[K,N] + bias[N]); A bf16, B fp32 converted on load.
//    Block tile 128x64, BK=32; 8 waves (wm 0..3, wn 0..1), each owns 32x32 =
//    2x2 v_wmma_f32_16x16x32_bf16. A tile staged via GLOBAL_LOAD_ASYNC_TO_LDS_B128;
//    B staged N-major (fragments = contiguous ds_load_b128 pairs). Tile t+1 is
//    in flight while tile t feeds the matrix pipe; one barrier per K-step.
template <bool RELU>
__global__ __launch_bounds__(256) void gemm_bf16_kernel(
    const unsigned short* __restrict__ A,   // MxK bf16
    const float*          __restrict__ Bw,  // KxN fp32 row-major
    const float*          __restrict__ bias,// N
    unsigned short*       __restrict__ Obf, // MxN bf16 (nullable)
    float*                __restrict__ Of32,// MxN fp32 (nullable)
    int M, int N, int K)
{
    __shared__ __bf16 As [2][128][40];  // [buf][m][k]  +8 pad; 80B rows (16B aligned)
    __shared__ __bf16 BsT[2][ 64][40];  // [buf][n][k]  transposed

    const int tid  = threadIdx.x;
    const int lane = tid & 31;
    const int wave = tid >> 5;
    const int wm   = wave >> 1;      // 0..3
    const int wn   = wave & 1;       // 0..1
    const int m0   = blockIdx.y * 128;
    const int n0   = blockIdx.x * 64;

    // B staging ownership: thread -> one column, 8 consecutive K values
    const int bc  = tid & 63;                          // column within tile
    const int bg  = tid >> 6;                          // 0..3 -> K rows bg*8..bg*8+7
    const bool cin = (n0 + bc) < N;
    const float* bsrc = Bw + (cin ? (n0 + bc) : 0);    // clamped, always in-bounds

    // A staging ownership: 2 x 16B chunks per thread
    const int ar0 = (tid * 8) >> 5,          ac0 = (tid * 8) & 31;
    const int ar1 = ((tid + 256) * 8) >> 5,  ac1 = ((tid + 256) * 8) & 31;

    // ---- staging helpers -------------------------------------------------
    auto stageA = [&](int k0, int buf) {
#if HAVE_ASYNC_LDS
        __builtin_amdgcn_global_load_async_to_lds_b128(
            ASYNC_GLB(A + (size_t)(m0 + ar0) * K + k0 + ac0),
            ASYNC_LDS(&As[buf][ar0][ac0]), 0, 0);
        __builtin_amdgcn_global_load_async_to_lds_b128(
            ASYNC_GLB(A + (size_t)(m0 + ar1) * K + k0 + ac1),
            ASYNC_LDS(&As[buf][ar1][ac1]), 0, 0);
#else
        *(uint4*)(&As[buf][ar0][ac0]) = *(const uint4*)(A + (size_t)(m0 + ar0) * K + k0 + ac0);
        *(uint4*)(&As[buf][ar1][ac1]) = *(const uint4*)(A + (size_t)(m0 + ar1) * K + k0 + ac1);
#endif
    };
    auto loadB = [&](int k0, float* r) {
        #pragma unroll
        for (int i = 0; i < 8; ++i)
            r[i] = bsrc[(size_t)(k0 + bg * 8 + i) * N];   // unconditional, clamped column
    };
    auto storeB = [&](const float* r, int buf) {
        unsigned pk[4];
        #pragma unroll
        for (int i = 0; i < 4; ++i)
            pk[i] = f2bf_pk(cin ? r[2 * i] : 0.0f, cin ? r[2 * i + 1] : 0.0f);
        *(uint4*)(&BsT[buf][bc][bg * 8]) = *(uint4*)pk;   // 16B aligned
    };

    v8f acc[2][2] = {};

    // ---- prologue: stage tile 0 ------------------------------------------
    stageA(0, 0);
    {
        float r[8];
        loadB(0, r);
        storeB(r, 0);
    }
#if HAVE_ASYNC_LDS
    __builtin_amdgcn_s_wait_asynccnt(0);
#endif
    __syncthreads();

    // ---- main pipeline ---------------------------------------------------
    for (int k0 = 0; k0 < K; k0 += 32) {
        const int  cur  = (k0 >> 5) & 1;
        const int  nxt  = cur ^ 1;
        const bool more = (k0 + 32) < K;

        float rB[8];
        if (more) {
            stageA(k0 + 32, nxt);     // ASYNCcnt-tracked DMA into next buffer
            loadB(k0 + 32, rB);       // coalesced fp32 loads for next B tile
            __builtin_prefetch(bsrc + (size_t)(k0 + 64) * N, 0, 0);  // two tiles ahead
        }

        // ---- fragments per ISA 7.12.2 VGPR layouts (contiguous 16B runs)
        v16bf af[2];
        #pragma unroll
        for (int i = 0; i < 2; ++i) {
            const int row = wm * 32 + i * 16 + (lane & 15);
            const int kb  = (lane >> 4) * 8;
            union { uint4 q[2]; v16bf v; } ua;
            ua.q[0] = *(const uint4*)(&As[cur][row][kb]);
            ua.q[1] = *(const uint4*)(&As[cur][row][16 + kb]);
            af[i] = ua.v;
        }
        v16bf bfm[2];
        #pragma unroll
        for (int j2 = 0; j2 < 2; ++j2) {
            const int col = wn * 32 + j2 * 16 + (lane & 15);
            const int kb  = (lane >> 4) * 8;
            union { uint4 q[2]; v16bf v; } ub;
            ub.q[0] = *(const uint4*)(&BsT[cur][col][kb]);
            ub.q[1] = *(const uint4*)(&BsT[cur][col][16 + kb]);
            bfm[j2] = ub.v;
        }

        #pragma unroll
        for (int i = 0; i < 2; ++i)
            #pragma unroll
            for (int j2 = 0; j2 < 2; ++j2)
                acc[i][j2] = __builtin_amdgcn_wmma_f32_16x16x32_bf16(
                    false, af[i], false, bfm[j2], (short)0, acc[i][j2], false, false);

        if (more) storeB(rB, nxt);    // convert + single ds_store_b128

#if HAVE_ASYNC_LDS
        __builtin_amdgcn_s_wait_asynccnt(0);
#endif
        __syncthreads();              // one barrier per K-step (double buffered)
    }

    // ---- epilogue: bias (+ReLU), C/D layout: M = e + 8*(lane>=16), N = lane&15
    #pragma unroll
    for (int i = 0; i < 2; ++i) {
        #pragma unroll
        for (int j2 = 0; j2 < 2; ++j2) {
            const int col = n0 + wn * 32 + j2 * 16 + (lane & 15);
            if (col >= N) continue;
            const float bz = bias[col];
            const int rbase = m0 + wm * 32 + i * 16 + ((lane >> 4) ? 8 : 0);
            #pragma unroll
            for (int e = 0; e < 8; ++e) {
                float v = acc[i][j2][e] + bz;
                if (RELU) v = fmaxf(v, 0.0f);
                const size_t idx = (size_t)(rbase + e) * N + col;
                if (Obf)  Obf[idx]  = f2bfu(v);
                if (Of32) Of32[idx] = v;
            }
        }
    }
}

// ---------------------------------------------------------------- 3) softmax + box decode + flatten
__global__ __launch_bounds__(128) void softmax_decode_kernel(
    const float* __restrict__ outbuf,   // d_out (logits at 0, bbox at OFF_BBOX)
    const float* __restrict__ props,
    const int*   __restrict__ imh,
    const int*   __restrict__ imw,
    float* __restrict__ flat_scores,
    float* __restrict__ flat_boxes)
{
    const int n = blockIdx.x;     // roi
    const int t = threadIdx.x;
    const int b = n / Pp;
    const int p = n % Pp;
    const float* lg = outbuf + OFF_LOGITS + (size_t)n * NCLS;

    __shared__ float sh[128];
    const float v = (t < NCLS) ? lg[t] : -FLT_MAX;
    sh[t] = v; __syncthreads();
    for (int s = 64; s > 0; s >>= 1) { if (t < s) sh[t] = fmaxf(sh[t], sh[t + s]); __syncthreads(); }
    const float mx = sh[0]; __syncthreads();
    sh[t] = (t < NCLS) ? __expf(v - mx) : 0.0f; __syncthreads();
    for (int s = 64; s > 0; s >>= 1) { if (t < s) sh[t] += sh[t + s]; __syncthreads(); }
    const float inv = 1.0f / sh[0];

    if (t < NCLS - 1) {
        const int c = t + 1;
        const float sc = __expf(lg[c] - mx) * inv;

        const float* dl = outbuf + OFF_BBOX + (size_t)n * (NCLS * 4) + c * 4;
        const float* pr = props + (size_t)n * 4;
        const float w  = pr[2] - pr[0], h = pr[3] - pr[1];
        const float cx = pr[0] + 0.5f * w, cy = pr[1] + 0.5f * h;
        const float dx = dl[0], dy = dl[1];
        const float dw = fminf(fmaxf(dl[2], -BBOX_CLIP), BBOX_CLIP);
        const float dh = fminf(fmaxf(dl[3], -BBOX_CLIP), BBOX_CLIP);
        const float pcx = dx * w + cx, pcy = dy * h + cy;
        const float pw = __expf(dw) * w, ph = __expf(dh) * h;
        const float fw = (float)imw[0], fh = (float)imh[0];

        const size_t q = (size_t)b * FLATN + (size_t)p * (NCLS - 1) + t;
        flat_scores[q] = sc;
        float* fb = flat_boxes + q * 4;
        fb[0] = fminf(fmaxf(pcx - 0.5f * pw, 0.0f), fw);
        fb[1] = fminf(fmaxf(pcy - 0.5f * ph, 0.0f), fh);
        fb[2] = fminf(fmaxf(pcx + 0.5f * pw, 0.0f), fw);
        fb[3] = fminf(fmaxf(pcy + 0.5f * ph, 0.0f), fh);
    }
}

// ---------------------------------------------------------------- 4) per-image top-1000 (hist select + bitonic sort)
__global__ __launch_bounds__(1024) void topk_kernel(
    const float* __restrict__ flat_scores,
    const float* __restrict__ flat_boxes,
    float* __restrict__ selS, int* __restrict__ selL, float* __restrict__ selB)
{
    __shared__ int   hist[4096];
    __shared__ float cs[2048];
    __shared__ int   ci[2048];
    __shared__ int   cnt, thr;

    const int img = blockIdx.x;
    const int t   = threadIdx.x;
    const float* sc = flat_scores + (size_t)img * FLATN;

    for (int i = t; i < 4096; i += 1024) hist[i] = 0;
    if (t == 0) cnt = 0;
    __syncthreads();

    for (int q = t; q < FLATN; q += 1024) {
        int b = (int)(sc[q] * 4096.0f);
        b = max(0, min(4095, b));
        atomicAdd(&hist[b], 1);
    }
    __syncthreads();

    if (t == 0) {
        int cum = 0, b = 4095;
        for (; b >= 0; --b) { cum += hist[b]; if (cum >= PRE_K) break; }
        thr = max(b, 0);
    }
    __syncthreads();
    const int T = thr;

    for (int q = t; q < FLATN; q += 1024) {
        const float s = sc[q];
        int b = (int)(s * 4096.0f);
        b = max(0, min(4095, b));
        if (b >= T) {
            const int p = atomicAdd(&cnt, 1);
            if (p < 2048) { cs[p] = s; ci[p] = q; }
        }
    }
    __syncthreads();
    const int c0 = min(cnt, 2048);
    for (int i = c0 + t; i < 2048; i += 1024) { cs[i] = -FLT_MAX; ci[i] = 0; }
    __syncthreads();

    // bitonic sort (descending), 2048 elems, 1024 threads x 2
    for (int k2 = 2; k2 <= 2048; k2 <<= 1) {
        for (int j = k2 >> 1; j > 0; j >>= 1) {
            #pragma unroll 1
            for (int e = 0; e < 2; ++e) {
                const int i   = t + e * 1024;
                const int ixj = i ^ j;
                if (ixj > i) {
                    const bool desc = ((i & k2) == 0);
                    const float a = cs[i], bq = cs[ixj];
                    const bool sw = desc ? (a < bq) : (a > bq);
                    if (sw) {
                        cs[i] = bq; cs[ixj] = a;
                        const int tmp = ci[i]; ci[i] = ci[ixj]; ci[ixj] = tmp;
                    }
                }
            }
            __syncthreads();
        }
    }

    if (t < PRE_K) {
        const float s = cs[t];
        const int   q = ci[t];
        const int lbl = 1 + (q % (NCLS - 1));
        selS[img * 1024 + t] = s;
        selL[img * 1024 + t] = lbl;
        const float* fb = flat_boxes + ((size_t)img * FLATN + q) * 4;
        float* sb = selB + ((size_t)img * 1024 + t) * 4;
        sb[0] = fb[0]; sb[1] = fb[1]; sb[2] = fb[2]; sb[3] = fb[3];
    }
}

// ---------------------------------------------------------------- 5) 1000x1000 IoU (class-offset boxes)
__global__ __launch_bounds__(256) void iou_kernel(
    const float* __restrict__ selB, const int* __restrict__ selL,
    float* __restrict__ iou)
{
    const int img = blockIdx.z;
    const int i = blockIdx.y * 16 + (threadIdx.x >> 4);
    const int j = blockIdx.x * 16 + (threadIdx.x & 15);
    if (i >= PRE_K || j >= PRE_K) return;

    const float* bi = selB + ((size_t)img * 1024 + i) * 4;
    const float* bj = selB + ((size_t)img * 1024 + j) * 4;
    const float oi = (float)selL[img * 1024 + i] * 10000.0f;
    const float oj = (float)selL[img * 1024 + j] * 10000.0f;

    const float ax1 = bi[0] + oi, ay1 = bi[1] + oi, ax2 = bi[2] + oi, ay2 = bi[3] + oi;
    const float bx1 = bj[0] + oj, by1 = bj[1] + oj, bx2 = bj[2] + oj, by2 = bj[3] + oj;
    const float areaA = (ax2 - ax1) * (ay2 - ay1);
    const float areaB = (bx2 - bx1) * (by2 - by1);
    const float iw = fmaxf(fminf(ax2, bx2) - fmaxf(ax1, bx1), 0.0f);
    const float ih = fmaxf(fminf(ay2, by2) - fmaxf(ay1, by1), 0.0f);
    const float inter = iw * ih;
    iou[((size_t)img * 1024 + i) * 1024 + j] = inter / (areaA + areaB - inter + 1e-9f);
}

// ---------------------------------------------------------------- 6) sequential NMS + top-100 emission
__global__ __launch_bounds__(1024) void nms_kernel(
    const float* __restrict__ selS, const int* __restrict__ selL,
    const float* __restrict__ selB, const float* __restrict__ iou,
    float* __restrict__ outbuf)
{
    __shared__ int keep[1024];
    const int img = blockIdx.x;
    const int t   = threadIdx.x;
    keep[t] = (t < PRE_K) && (selS[img * 1024 + t] > SCORE_T);
    __syncthreads();

    const float* IR = iou + (size_t)img * 1024 * 1024;
    for (int i = 0; i < PRE_K; ++i) {
        const int ki = keep[i];
        if (ki && t > i && t < PRE_K && keep[t] && IR[(size_t)i * 1024 + t] > NMS_T)
            keep[t] = 0;
        __syncthreads();
    }

    if (t == 0) {
        float* db = outbuf + OFF_DETB + (size_t)img * DETN * 4;
        float* ds = outbuf + OFF_DETS + (size_t)img * DETN;
        float* dl = outbuf + OFF_DETL + (size_t)img * DETN;
        int w = 0;
        for (int i = 0; i < PRE_K && w < DETN; ++i) {
            if (keep[i]) {
                const float* sb = selB + ((size_t)img * 1024 + i) * 4;
                db[w * 4 + 0] = sb[0]; db[w * 4 + 1] = sb[1];
                db[w * 4 + 2] = sb[2]; db[w * 4 + 3] = sb[3];
                ds[w] = selS[img * 1024 + i];
                dl[w] = (float)selL[img * 1024 + i];
                ++w;
            }
        }
        for (int i = 0; i < PRE_K && w < DETN; ++i) {
            if (!keep[i]) {
                const float* sb = selB + ((size_t)img * 1024 + i) * 4;
                db[w * 4 + 0] = sb[0]; db[w * 4 + 1] = sb[1];
                db[w * 4 + 2] = sb[2]; db[w * 4 + 3] = sb[3];
                ds[w] = -1.0f;
                dl[w] = (float)selL[img * 1024 + i];
                ++w;
            }
        }
    }
}

// ---------------------------------------------------------------- launch
extern "C" void kernel_launch(void* const* d_in, const int* in_sizes, int n_in,
                              void* d_out, int out_size, void* d_ws, size_t ws_size,
                              hipStream_t stream) {
    (void)in_sizes; (void)n_in; (void)out_size; (void)ws_size;

    const float* features  = (const float*)d_in[0];
    const float* proposals = (const float*)d_in[1];
    const float* fc6_w  = (const float*)d_in[2];
    const float* fc6_b  = (const float*)d_in[3];
    const float* fc7_w  = (const float*)d_in[4];
    const float* fc7_b  = (const float*)d_in[5];
    const float* cls_w  = (const float*)d_in[6];
    const float* cls_b  = (const float*)d_in[7];
    const float* bbox_w = (const float*)d_in[8];
    const float* bbox_b = (const float*)d_in[9];
    const int*   imh    = (const int*)d_in[10];
    const int*   imw    = (const int*)d_in[11];

    float* out = (float*)d_out;
    char*  ws  = (char*)d_ws;
    unsigned short* X    = (unsigned short*)(ws + WS_X);
    unsigned short* H6   = (unsigned short*)(ws + WS_H6);
    unsigned short* H7   = (unsigned short*)(ws + WS_H7);
    float* flat_scores   = (float*)(ws + WS_FSC);
    float* flat_boxes    = (float*)(ws + WS_FBX);
    float* selS          = (float*)(ws + WS_SELS);
    int*   selL          = (int*)  (ws + WS_SELL);
    float* selB          = (float*)(ws + WS_SELB);
    float* iou           = (float*)(ws + WS_IOU);

    // 1) ROI-Align -> bf16 activations
    roi_align_kernel<<<NROI, 256, 0, stream>>>(features, proposals, X);

    // 2) FC6: (1024 x 12544) @ (12544 x 4096) + relu -> bf16
    gemm_bf16_kernel<true><<<dim3(REP / 64, NROI / 128), 256, 0, stream>>>(
        X, fc6_w, fc6_b, H6, nullptr, NROI, REP, D_IN);

    // 3) FC7: (1024 x 4096) @ (4096 x 4096) + relu -> bf16
    gemm_bf16_kernel<true><<<dim3(REP / 64, NROI / 128), 256, 0, stream>>>(
        H6, fc7_w, fc7_b, H7, nullptr, NROI, REP, REP);

    // 4) cls logits -> d_out[0 .. 1024*81)
    gemm_bf16_kernel<false><<<dim3((NCLS + 63) / 64, NROI / 128), 256, 0, stream>>>(
        H7, cls_w, cls_b, nullptr, out + OFF_LOGITS, NROI, NCLS, REP);

    // 5) bbox regression -> d_out[OFF_BBOX ..)
    gemm_bf16_kernel<false><<<dim3((NCLS * 4 + 63) / 64, NROI / 128), 256, 0, stream>>>(
        H7, bbox_w, bbox_b, nullptr, out + OFF_BBOX, NROI, NCLS * 4, REP);

    // 6) softmax + decode + flatten
    softmax_decode_kernel<<<NROI, 128, 0, stream>>>(
        out, proposals, imh, imw, flat_scores, flat_boxes);

    // 7) per-image top-1000
    topk_kernel<<<Bn, 1024, 0, stream>>>(flat_scores, flat_boxes, selS, selL, selB);

    // 8) IoU matrix
    iou_kernel<<<dim3((PRE_K + 15) / 16, (PRE_K + 15) / 16, Bn), 256, 0, stream>>>(
        selB, selL, iou);

    // 9) NMS + detection outputs
    nms_kernel<<<Bn, 1024, 0, stream>>>(selS, selL, selB, iou, out);
}
